// GCGRU_9380208574542
// MI455X (gfx1250) — matrix operand order repaired and checked by reading, not compiled, not run on hardware
//
#include <hip/hip_runtime.h>
#include <math.h>

// GCGRU (DCRNN-style) forward for MI455X gfx1250.
// bf16 WMMA (v_wmma_f32_16x16x32_bf16) for both GEMM stages of every
// graph-conv call; f32 accumulate; channel dims zero-padded to multiples of
// 16. GEMM: 64xNcols block tile, 8 waves as 2(M)x4(N) groups, each wave a
// 2x2 grid of 16x16 accumulators; double-buffered LDS with b128 staging.

typedef __attribute__((ext_vector_type(16))) __bf16 v16bf;
typedef __attribute__((ext_vector_type(8)))  float  v8f;

namespace {
constexpr int NNODE = 1024;
constexpr int NB    = 64;
constexpr int NT    = 12;
constexpr int NHOR  = 12;
constexpr int NC    = 2;
constexpr int NH    = 64;
constexpr int NK    = 6;          // CHEB * NSUP
constexpr int PP0   = 80;         // pad(C+H=66) -> 80
constexpr int PP1   = 128;        // H+H
constexpr int KQ0   = NK * PP0;   // 480
constexpr int KQ1   = NK * PP1;   // 768
}

__device__ __forceinline__ unsigned short f2bf(float f) {
  unsigned int u = __float_as_uint(f);
  u += 0x7FFFu + ((u >> 16) & 1u);   // round-to-nearest-even
  return (unsigned short)(u >> 16);
}

union FragBf { v16bf v; unsigned int u[8]; };
union U4     { uint4 v; unsigned int u[4]; unsigned short s[8]; };

// ---------------------------------------------------------------------------
// Generic bf16 WMMA GEMM:  C = A(MxKd, row-major, lda) * B(KdxNcols, ldb)
// Block = 256 threads = 8 waves; tile = 64 rows x Ncols (Ncols in {64,80,128},
// M % 64 == 0, Kd % 32 == 0, lda/ldb multiples of 8).
// blockIdx.z indexes (k,b) batch pairs:
//   k = z % kcount, b = z / kcount
//   A += k*aStrideK; B += b*bStrideB; C += b*cStrideB + k*cOffK
// outBf16: store bf16 (graph-conv -> S) else f32 (pre-activations).
// ---------------------------------------------------------------------------
__global__ __launch_bounds__(256) void wmma_gemm_bf16(
    const unsigned short* __restrict__ Abase,
    const unsigned short* __restrict__ Bbase,
    void* __restrict__ Cbase,
    int M, int Kd, int Ncols,
    int lda, int ldb, int ldc,
    int kcount,
    long long aStrideK, long long bStrideB, long long cStrideB, long long cOffK,
    int outBf16)
{
  (void)M;
  const int z  = (int)blockIdx.z;
  const int kb = z % kcount;
  const int bb = z / kcount;
  const unsigned short* __restrict__ A  = Abase + (long long)kb * aStrideK;
  const unsigned short* __restrict__ Bm = Bbase + (long long)bb * bStrideB;
  const long long cOff = (long long)bb * cStrideB + (long long)kb * cOffK;

  const int m0   = (int)blockIdx.x * 64;
  const int tid  = (int)threadIdx.x;
  const int lane = tid & 31;
  // wave-group ids forced scalar so WMMA guards are uniform s_cbranch,
  // never EXEC-mask manipulation (WMMA requires EXEC all-ones).
  const int wmg  = __builtin_amdgcn_readfirstlane((tid >> 5) & 1);  // M group
  const int wng  = __builtin_amdgcn_readfirstlane(tid >> 6);        // N group 0..3
  const int CT   = Ncols >> 4;
  const int hf   = lane >> 4;
  const int lr   = lane & 15;
  const bool do1 = (wng + 4) < CT;     // second N tile valid (scalar)

  // LDS: double-buffered A (64x32) and transposed B ([n][k], 128x32 max).
  // Row stride 40 halfwords (80 B): 16-B aligned b128 staging for A and
  // conflict-free fragment reads (20-dword lane stride vs 64 banks).
  __shared__ unsigned short At[2][64][40];
  __shared__ unsigned short Bt[2][128][40];

  v8f acc00 = {}, acc01 = {}, acc10 = {}, acc11 = {};

  const int ncu   = Ncols >> 1;        // dwords per B row
  const int nq4   = ncu >> 2;          // uint4 per B row
  const int bTot4 = 32 * nq4;          // uint4 per B tile

  const int aRow = tid >> 2;           // 0..63
  const int aQ   = (tid & 3) << 3;     // element offset in row (0,8,16,24)

  U4 aReg; U4 bReg[2];
  int bRow[2] = {0, 0}, bN0[2] = {0, 0};
  bool bVal[2] = {false, false};

  auto loadTile = [&](int k0) {
    aReg.v = *(const uint4*)(&A[(long long)(m0 + aRow) * lda + (k0 + aQ)]);
#pragma unroll
    for (int i = 0; i < 2; ++i) {
      int e = tid + i * 256;
      bVal[i] = e < bTot4;
      if (bVal[i]) {
        int row = e / nq4;
        int qq  = e - row * nq4;
        bRow[i] = row;
        bN0[i]  = qq << 3;
        bReg[i].v = *(const uint4*)(&Bm[(long long)(k0 + row) * ldb + (qq << 3)]);
      }
    }
  };
  auto storeTile = [&](int buf) {
    *(uint4*)(&At[buf][aRow][aQ]) = aReg.v;
#pragma unroll
    for (int i = 0; i < 2; ++i) {
      if (bVal[i]) {
#pragma unroll
        for (int j = 0; j < 8; ++j)
          Bt[buf][bN0[i] + j][bRow[i]] = bReg[i].s[j];
      }
    }
  };

  loadTile(0);
  storeTile(0);
  int cur = 0;

  for (int k0 = 0; k0 < Kd; k0 += 32) {
    __syncthreads();
    const bool hasNext = (k0 + 32) < Kd;
    if (hasNext) loadTile(k0 + 32);    // global b128 prefetch overlaps WMMA

    // A fragments: lane=row, VGPR r -> K = (r/4)*16 + half*8 + (r%4)*2
    FragBf af0, af1;
#pragma unroll
    for (int r = 0; r < 8; ++r) {
      int kk = ((r >> 2) << 4) + (hf << 3) + ((r & 3) << 1);
      af0.u[r] = *(const unsigned int*)(&At[cur][(wmg * 2 + 0) * 16 + lr][kk]);
      af1.u[r] = *(const unsigned int*)(&At[cur][(wmg * 2 + 1) * 16 + lr][kk]);
    }
    // B fragment 0: lane=column, VGPR r -> K = (r/4)*16 + half*8 + (r%4)*2
    {
      FragBf bf0;
#pragma unroll
      for (int r = 0; r < 8; ++r) {
        int kk = ((r >> 2) << 4) + (hf << 3) + ((r & 3) << 1);
        bf0.u[r] = *(const unsigned int*)(&Bt[cur][wng * 16 + lr][kk]);
      }
      acc00 = __builtin_amdgcn_wmma_f32_16x16x32_bf16(
          false, af0.v, false, bf0.v, (short)0, acc00, false, false);
      acc10 = __builtin_amdgcn_wmma_f32_16x16x32_bf16(
          false, af1.v, false, bf0.v, (short)0, acc10, false, false);
    }
    if (do1) {
      FragBf bf1;
#pragma unroll
      for (int r = 0; r < 8; ++r) {
        int kk = ((r >> 2) << 4) + (hf << 3) + ((r & 3) << 1);
        bf1.u[r] = *(const unsigned int*)(&Bt[cur][(wng + 4) * 16 + lr][kk]);
      }
      acc01 = __builtin_amdgcn_wmma_f32_16x16x32_bf16(
          false, af0.v, false, bf1.v, (short)0, acc01, false, false);
      acc11 = __builtin_amdgcn_wmma_f32_16x16x32_bf16(
          false, af1.v, false, bf1.v, (short)0, acc11, false, false);
    }
    if (hasNext) storeTile(cur ^ 1);
    cur ^= 1;
  }

  // C layout: lane=column (lr), half hf, VGPR r -> row = 8*hf + r
  auto storeAcc = [&](v8f& acc, int mt, int c) {
    int col = c * 16 + lr;
#pragma unroll
    for (int r = 0; r < 8; ++r) {
      long long row = (long long)(m0 + mt * 16 + hf * 8 + r);
      long long idx = cOff + row * (long long)ldc + col;
      if (outBf16) ((unsigned short*)Cbase)[idx] = f2bf(acc[r]);
      else         ((float*)Cbase)[idx] = acc[r];
    }
  };
  storeAcc(acc00, wmg * 2 + 0, wng);
  storeAcc(acc10, wmg * 2 + 1, wng);
  if (do1) {
    storeAcc(acc01, wmg * 2 + 0, wng + 4);
    storeAcc(acc11, wmg * 2 + 1, wng + 4);
  }
}

// ---------------------------------------------------------------------------
// Elementwise / setup kernels
// ---------------------------------------------------------------------------
__global__ void k_f32_to_bf16(const float* __restrict__ in,
                              unsigned short* __restrict__ out, int total) {
  int i = blockIdx.x * blockDim.x + threadIdx.x;
  if (i < total) out[i] = f2bf(in[i]);
}

// W (K*P x Ncols) f32 -> Wp (K*Pp x Ncols) bf16, zero rows for p in [P,Pp)
__global__ void k_repack_w(const float* __restrict__ W,
                           unsigned short* __restrict__ Wp,
                           int P, int Pp, int Ncols, int total) {
  int i = blockIdx.x * blockDim.x + threadIdx.x;
  if (i >= total) return;
  int q  = i / Ncols;
  int nn = i - q * Ncols;
  int k  = q / Pp;
  int p  = q - k * Pp;
  float v = (p < P) ? W[(long long)(k * P + p) * Ncols + nn] : 0.f;
  Wp[i] = f2bf(v);
}

// comb[b,n,p] = p<Cin ? xin[b,n,p] : p<Cin+H ? h[b,n,p-Cin] : 0   (bf16)
__global__ void k_build_comb(unsigned short* __restrict__ comb,
                             const float* __restrict__ xin, long long xinBStride,
                             int Cin, const float* __restrict__ h,
                             int Pp, int total) {
  int i = blockIdx.x * blockDim.x + threadIdx.x;
  if (i >= total) return;
  int p  = i % Pp;
  int bn = i / Pp;
  int b  = bn / NNODE;
  int n  = bn - b * NNODE;
  float v = 0.f;
  if (p < Cin)            v = xin[(long long)b * xinBStride + (long long)n * Cin + p];
  else if (p < Cin + NH)  v = h[(long long)bn * NH + (p - Cin)];
  comb[i] = f2bf(v);
}

// zr = sigmoid(pre + gb); z -> zbuf; comb2 = concat(xin, r*h) (bf16, padded)
__global__ void k_gates_comb2(const float* __restrict__ pre,
                              const float* __restrict__ gb,
                              const float* __restrict__ h,
                              const float* __restrict__ xin, long long xinBStride,
                              int Cin,
                              float* __restrict__ zbuf,
                              unsigned short* __restrict__ comb2,
                              int Pp, int total) {
  int i = blockIdx.x * blockDim.x + threadIdx.x;
  if (i >= total) return;
  int p  = i % Pp;
  int bn = i / Pp;
  int b  = bn / NNODE;
  int n  = bn - b * NNODE;
  float v = 0.f;
  if (p < Cin) {
    v = xin[(long long)b * xinBStride + (long long)n * Cin + p];
  } else if (p < Cin + NH) {
    int hh = p - Cin;
    float zv = 1.f / (1.f + expf(-(pre[(long long)bn * (2 * NH) + hh] + gb[hh])));
    float rv = 1.f / (1.f + expf(-(pre[(long long)bn * (2 * NH) + NH + hh] + gb[NH + hh])));
    zbuf[(long long)bn * NH + hh] = zv;
    v = rv * h[(long long)bn * NH + hh];
  }
  comb2[i] = f2bf(v);
}

// h = z*tanh(pre+ub) + (1-z)*h   (in place)
__global__ void k_gru_update(const float* __restrict__ pre,
                             const float* __restrict__ ub,
                             const float* __restrict__ zbuf,
                             float* __restrict__ h, int total) {
  int i = blockIdx.x * blockDim.x + threadIdx.x;
  if (i >= total) return;
  int hh = i % NH;
  float nv = tanhf(pre[i] + ub[hh]);
  float z  = zbuf[i];
  h[i] = z * nv + (1.f - z) * h[i];
}

// out[b,t,n,c] = h1[b,n,:] @ projW[:,c] + projb[c];  also feed back dec_in
__global__ void k_proj_out(const float* __restrict__ h1,
                           const float* __restrict__ projW,
                           const float* __restrict__ projb,
                           float* __restrict__ out,
                           float* __restrict__ dec_in,
                           int t, int total) {
  int i = blockIdx.x * blockDim.x + threadIdx.x;
  if (i >= total) return;
  int c  = i % NC;
  int bn = i / NC;
  int b  = bn / NNODE;
  int n  = bn - b * NNODE;
  float acc = projb[c];
#pragma unroll
  for (int hh = 0; hh < NH; ++hh)
    acc += h1[(long long)bn * NH + hh] * projW[hh * NC + c];
  dec_in[i] = acc;
  out[((long long)b * NHOR + t) * (NNODE * NC) + (long long)n * NC + c] = acc;
}

// ---------------------------------------------------------------------------
// Host-side: one GRU cell = build comb -> gconv -> gates GEMM -> gates/comb2
//            -> gconv -> update GEMM -> GRU update
// ---------------------------------------------------------------------------
static void run_cell(hipStream_t stream,
                     const unsigned short* G_bf,
                     const float* xin, long long xinBStride, int Cin,
                     int Pp, int Kq,
                     const unsigned short* Wg, const float* gb,
                     const unsigned short* Wu, const float* ub,
                     float* h,
                     unsigned short* comb, unsigned short* S,
                     float* preact, float* zbuf)
{
  const int totC = NB * NNODE * Pp;
  const int totH = NB * NNODE * NH;
  dim3 gConv(NNODE / 64, 1, NB * NK);
  dim3 gDense((NB * NNODE) / 64, 1, 1);

  k_build_comb<<<(totC + 255) / 256, 256, 0, stream>>>(comb, xin, xinBStride, Cin, h, Pp, totC);

  // S[b, i, k*Pp+p] = sum_j G[k,i,j] * comb[b,j,p]   (bf16 out)
  wmma_gemm_bf16<<<gConv, 256, 0, stream>>>(
      G_bf, comb, S, NNODE, NNODE, Pp, NNODE, Pp, Kq,
      NK, (long long)NNODE * NNODE, (long long)NNODE * Pp,
      (long long)NNODE * Kq, (long long)Pp, 1);

  // gates pre-activation: (B*N x Kq) @ (Kq x 2H) -> f32
  wmma_gemm_bf16<<<gDense, 256, 0, stream>>>(
      S, Wg, preact, NB * NNODE, Kq, 2 * NH, Kq, 2 * NH, 2 * NH,
      1, 0, 0, 0, 0, 0);

  k_gates_comb2<<<(totC + 255) / 256, 256, 0, stream>>>(
      preact, gb, h, xin, xinBStride, Cin, zbuf, comb, Pp, totC);

  wmma_gemm_bf16<<<gConv, 256, 0, stream>>>(
      G_bf, comb, S, NNODE, NNODE, Pp, NNODE, Pp, Kq,
      NK, (long long)NNODE * NNODE, (long long)NNODE * Pp,
      (long long)NNODE * Kq, (long long)Pp, 1);

  // update pre-activation: (B*N x Kq) @ (Kq x H) -> f32
  wmma_gemm_bf16<<<gDense, 256, 0, stream>>>(
      S, Wu, preact, NB * NNODE, Kq, NH, Kq, NH, NH,
      1, 0, 0, 0, 0, 0);

  k_gru_update<<<(totH + 255) / 256, 256, 0, stream>>>(preact, ub, zbuf, h, totH);
}

extern "C" void kernel_launch(void* const* d_in, const int* in_sizes, int n_in,
                              void* d_out, int out_size, void* d_ws, size_t ws_size,
                              hipStream_t stream)
{
  (void)in_sizes; (void)n_in; (void)out_size; (void)ws_size;

  const float* x = (const float*)d_in[0];
  const float* G = (const float*)d_in[1];
  // layer order: 0=enc0, 1=enc1, 2=dec0, 3=dec1
  const float* wGW[4] = { (const float*)d_in[2],  (const float*)d_in[6],
                          (const float*)d_in[10], (const float*)d_in[14] };
  const float* wGB[4] = { (const float*)d_in[3],  (const float*)d_in[7],
                          (const float*)d_in[11], (const float*)d_in[15] };
  const float* wUW[4] = { (const float*)d_in[4],  (const float*)d_in[8],
                          (const float*)d_in[12], (const float*)d_in[16] };
  const float* wUB[4] = { (const float*)d_in[5],  (const float*)d_in[9],
                          (const float*)d_in[13], (const float*)d_in[17] };
  const float* projW = (const float*)d_in[18];
  const float* projb = (const float*)d_in[19];
  float* out = (float*)d_out;

  const int Pcur[4]  = { NC + NH, NH + NH, NC + NH, NH + NH };   // 66,128,66,128
  const int Ppad[4]  = { PP0, PP1, PP0, PP1 };
  const int Kq[4]    = { KQ0, KQ1, KQ0, KQ1 };

  // ---- carve workspace ----
  char* wp = (char*)d_ws;
  auto carve = [&](size_t bytes) -> char* {
    char* p = wp;
    wp += (bytes + 255) & ~((size_t)255);
    return p;
  };
  unsigned short* G_bf = (unsigned short*)carve((size_t)NK * NNODE * NNODE * 2);
  unsigned short* Wg_bf[4];
  unsigned short* Wu_bf[4];
  for (int l = 0; l < 4; ++l) {
    Wg_bf[l] = (unsigned short*)carve((size_t)Kq[l] * 2 * NH * 2);
    Wu_bf[l] = (unsigned short*)carve((size_t)Kq[l] * NH * 2);
  }
  unsigned short* comb = (unsigned short*)carve((size_t)NB * NNODE * PP1 * 2);
  unsigned short* S    = (unsigned short*)carve((size_t)NB * NNODE * KQ1 * 2);
  float* preact = (float*)carve((size_t)NB * NNODE * 2 * NH * 4);
  float* zbuf   = (float*)carve((size_t)NB * NNODE * NH * 4);
  float* h0     = (float*)carve((size_t)NB * NNODE * NH * 4);
  float* h1     = (float*)carve((size_t)NB * NNODE * NH * 4);
  float* dec_in = (float*)carve((size_t)NB * NNODE * NC * 4);

  // ---- per-call conversions (deterministic, no caching) ----
  {
    int tot = NK * NNODE * NNODE;
    k_f32_to_bf16<<<(tot + 255) / 256, 256, 0, stream>>>(G, G_bf, tot);
  }
  for (int l = 0; l < 4; ++l) {
    int totg = NK * Ppad[l] * 2 * NH;
    k_repack_w<<<(totg + 255) / 256, 256, 0, stream>>>(
        wGW[l], Wg_bf[l], Pcur[l], Ppad[l], 2 * NH, totg);
    int totu = NK * Ppad[l] * NH;
    k_repack_w<<<(totu + 255) / 256, 256, 0, stream>>>(
        wUW[l], Wu_bf[l], Pcur[l], Ppad[l], NH, totu);
  }
  (void)hipMemsetAsync(h0, 0, (size_t)NB * NNODE * NH * 4, stream);
  (void)hipMemsetAsync(h1, 0, (size_t)NB * NNODE * NH * 4, stream);
  (void)hipMemsetAsync(dec_in, 0, (size_t)NB * NNODE * NC * 4, stream);

  // ---- encoder: stacked-GRU scan, time-major ----
  for (int t = 0; t < NT; ++t) {
    run_cell(stream, G_bf, x + (size_t)t * NNODE * NC, (long long)NT * NNODE * NC,
             NC, PP0, KQ0, Wg_bf[0], wGB[0], Wu_bf[0], wUB[0],
             h0, comb, S, preact, zbuf);
    run_cell(stream, G_bf, h0, (long long)NNODE * NH,
             NH, PP1, KQ1, Wg_bf[1], wGB[1], Wu_bf[1], wUB[1],
             h1, comb, S, preact, zbuf);
  }

  // ---- decoder: h0/h1 carry encoder finals; out fed back ----
  for (int t = 0; t < NHOR; ++t) {
    run_cell(stream, G_bf, dec_in, (long long)NNODE * NC,
             NC, PP0, KQ0, Wg_bf[2], wGB[2], Wu_bf[2], wUB[2],
             h0, comb, S, preact, zbuf);
    run_cell(stream, G_bf, h0, (long long)NNODE * NH,
             NH, PP1, KQ1, Wg_bf[3], wGB[3], Wu_bf[3], wUB[3],
             h1, comb, S, preact, zbuf);
    int tot = NB * NNODE * NC;
    k_proj_out<<<(tot + 255) / 256, 256, 0, stream>>>(
        h1, projW, projb, out, dec_in, t, tot);
  }
}
